// InterpolationNet_87170656239959
// MI455X (gfx1250) — matrix operand order repaired and checked by reading, not compiled, not run on hardware
//
#include <hip/hip_runtime.h>
#include <math.h>

typedef __attribute__((ext_vector_type(16))) _Float16 v16h;
typedef __attribute__((ext_vector_type(8)))  _Float16 v8h;
typedef __attribute__((ext_vector_type(8)))  float    v8f;

__device__ __forceinline__ int iclampd(int v, int lo, int hi) {
    return v < lo ? lo : (v > hi ? hi : v);
}

// ---------------------------------------------------------------------------
// Weight repack: (cout, cin, 3, 3) f32 OIHW  ->  f16 [tap][coutP][cinP]
// Zero-padded so conv kernel never needs channel guards.
// ---------------------------------------------------------------------------
__global__ void prep_w_kernel(const float* __restrict__ w, _Float16* __restrict__ dst,
                              int cin, int cout, int cinP, int coutP)
{
    int i = blockIdx.x * blockDim.x + threadIdx.x;
    int tot = 9 * coutP * cinP;
    if (i >= tot) return;
    int c   = i % cinP;
    int t   = i / cinP;
    int co  = t % coutP;
    int tap = t / coutP;
    float v = 0.f;
    if (c < cin && co < cout) v = w[((size_t)co * cin + c) * 9 + tap];
    dst[((size_t)(tap * coutP + co)) * cinP + c] = (_Float16)v;
}

// ---------------------------------------------------------------------------
// Pack concat(frame_prev, frame_next): f32 NCHW -> f16 NHWC padded to 32 ch
// ---------------------------------------------------------------------------
__global__ void pack_in_kernel(const float* __restrict__ fp, const float* __restrict__ fn,
                               _Float16* __restrict__ dst, int hw, int Bn)
{
    int i = blockIdx.x * blockDim.x + threadIdx.x;
    int tot = Bn * hw * 32;
    if (i >= tot) return;
    int c = i & 31;
    int p = i >> 5;
    int b = p / hw;
    int r = p - b * hw;
    float v = 0.f;
    if (c < 3)      v = fp[((size_t)b * 3 + c) * hw + r];
    else if (c < 6) v = fn[((size_t)b * 3 + (c - 3)) * hw + r];
    dst[i] = (_Float16)v;
}

// ---------------------------------------------------------------------------
// Implicit-GEMM 3x3 conv, WMMA f32 = f16 x f16 + f32, register-blocked 2x2:
// each wave computes a 32-pixel x 32-cout tile (4 accumulators), so one
// K=32 chunk costs 2 A-fragment loads + 2 B-fragment loads for 4 WMMAs.
// Activations NHWC f16 (cinP multiple of 32); weights [tap][coutP][cinP] f16.
// 8 waves/block => 256 pixels per block. grid.y covers coutP/32 groups.
// Epilogue: +bias, optional ReLU, optional residual add (relu(conv)+skip).
// ---------------------------------------------------------------------------
__global__ void __launch_bounds__(256)
conv3x3_wmma_kernel(const _Float16* __restrict__ in,
                    const _Float16* __restrict__ wgt,
                    const float* __restrict__ bias, int biasN,
                    const _Float16* __restrict__ resid,
                    _Float16* __restrict__ out,
                    int total, int H, int W, int cinP, int coutP, int relu)
{
    const int lane  = threadIdx.x & 31;
    const int wave  = threadIdx.x >> 5;
    const int g     = blockIdx.y;              // 32-cout group
    const int mBase = (blockIdx.x * 8 + wave) * 32;
    const int mrow  = lane & 15;               // A row M / B column N
    const int khalf = lane >> 4;               // K half-group held by this lane

    const int hw = H * W;

    // Two M-tiles of 16 pixels each.
    int py[2], px[2], pb[2], pix[2];
#pragma unroll
    for (int mb = 0; mb < 2; ++mb) {
        int p  = mBase + mb * 16 + mrow;
        pix[mb] = p;
        int pc = p < total ? p : total - 1;
        int bi = pc / hw;
        int rem = pc - bi * hw;
        pb[mb] = bi;
        py[mb] = rem / W;
        px[mb] = rem - py[mb] * W;
    }

    const v8h zero8 = {};
    v8f acc[2][2] = { { {}, {} }, { {}, {} } };

    for (int tap = 0; tap < 9; ++tap) {
        const int dy = tap / 3 - 1;
        const int dx = tap - (tap / 3) * 3 - 1;

        const _Float16* abase[2];
        bool av[2];
#pragma unroll
        for (int mb = 0; mb < 2; ++mb) {
            int ny = py[mb] + dy, nx = px[mb] + dx;
            av[mb] = (pix[mb] < total) & ((unsigned)ny < (unsigned)H)
                                       & ((unsigned)nx < (unsigned)W);
            int nyc = iclampd(ny, 0, H - 1);
            int nxc = iclampd(nx, 0, W - 1);
            abase[mb] = in + ((size_t)(pb[mb] * H + nyc) * W + nxc) * (size_t)cinP
                           + khalf * 16;
        }
        // B rows for the two 16-cout groups (N = mrow within each group).
        const _Float16* brow0 = wgt + ((size_t)(tap * coutP + g * 32 + mrow)) * (size_t)cinP
                                    + khalf * 16;
        const _Float16* brow1 = brow0 + (size_t)16 * cinP;

        for (int c0 = 0; c0 < cinP; c0 += 32) {
            v16h Af[2];
#pragma unroll
            for (int mb = 0; mb < 2; ++mb) {
                v8h a0 = *(const v8h*)(abase[mb] + c0);
                v8h a1 = *(const v8h*)(abase[mb] + c0 + 8);
                a0 = av[mb] ? a0 : zero8;          // cndmask, no EXEC change
                a1 = av[mb] ? a1 : zero8;
#pragma unroll
                for (int k = 0; k < 8; ++k) { Af[mb][k] = a0[k]; Af[mb][k + 8] = a1[k]; }
            }
            v16h B0 = *(const v16h*)(brow0 + c0);
            v16h B1 = *(const v16h*)(brow1 + c0);
            acc[0][0] = __builtin_amdgcn_wmma_f32_16x16x32_f16(
                false, Af[0], false, B0, (short)0, acc[0][0], false, false);
            acc[0][1] = __builtin_amdgcn_wmma_f32_16x16x32_f16(
                false, Af[0], false, B1, (short)0, acc[0][1], false, false);
            acc[1][0] = __builtin_amdgcn_wmma_f32_16x16x32_f16(
                false, Af[1], false, B0, (short)0, acc[1][0], false, false);
            acc[1][1] = __builtin_amdgcn_wmma_f32_16x16x32_f16(
                false, Af[1], false, B1, (short)0, acc[1][1], false, false);
        }
    }

#pragma unroll
    for (int nb = 0; nb < 2; ++nb) {
        const int n = g * 32 + nb * 16 + mrow;
        float bv = (n < biasN) ? bias[n] : 0.f;
#pragma unroll
        for (int mb = 0; mb < 2; ++mb) {
#pragma unroll
            for (int v = 0; v < 8; ++v) {
                int q = mBase + mb * 16 + khalf * 8 + v;   // C/D: M = v + 8*khalf
                if (q < total) {
                    float r = acc[mb][nb][v] + bv;
                    if (relu) r = fmaxf(r, 0.f);
                    size_t oi = (size_t)q * (size_t)coutP + n;
                    if (resid) r += (float)resid[oi];
                    out[oi] = (_Float16)r;
                }
            }
        }
    }
}

// ---------------------------------------------------------------------------
// 2x2 average pool, NHWC f16
// ---------------------------------------------------------------------------
__global__ void pool2_kernel(const _Float16* __restrict__ in, _Float16* __restrict__ out,
                             int Bn, int Hi, int Wi, int C)
{
    int Ho = Hi >> 1, Wo = Wi >> 1;
    int i = blockIdx.x * blockDim.x + threadIdx.x;
    int tot = Bn * Ho * Wo * C;
    if (i >= tot) return;
    int c = i % C;
    int p = i / C;
    int b = p / (Ho * Wo);
    int r = p - b * Ho * Wo;
    int oy = r / Wo, ox = r - oy * Wo;
    size_t base = ((size_t)(b * Hi + 2 * oy) * Wi + 2 * ox) * (size_t)C + c;
    size_t rs = (size_t)Wi * C;
    float s = (float)in[base] + (float)in[base + C]
            + (float)in[base + rs] + (float)in[base + rs + C];
    out[i] = (_Float16)(0.25f * s);
}

// ---------------------------------------------------------------------------
// Bilinear 2x upsample (half-pixel centers, edge clamp), NHWC f16
// ---------------------------------------------------------------------------
__global__ void up2_kernel(const _Float16* __restrict__ in, _Float16* __restrict__ out,
                           int Bn, int Hi, int Wi, int C)
{
    int Ho = Hi * 2, Wo = Wi * 2;
    int i = blockIdx.x * blockDim.x + threadIdx.x;
    int tot = Bn * Ho * Wo * C;
    if (i >= tot) return;
    int c = i % C;
    int p = i / C;
    int b = p / (Ho * Wo);
    int r = p - b * Ho * Wo;
    int oy = r / Wo, ox = r - oy * Wo;
    float sy = oy * 0.5f - 0.25f;
    float sx = ox * 0.5f - 0.25f;
    int y0 = (int)floorf(sy), x0 = (int)floorf(sx);
    float wy = sy - y0, wx = sx - x0;
    int y0c = iclampd(y0, 0, Hi - 1), y1c = iclampd(y0 + 1, 0, Hi - 1);
    int x0c = iclampd(x0, 0, Wi - 1), x1c = iclampd(x0 + 1, 0, Wi - 1);
    auto ld = [&](int yy, int xx) {
        return (float)in[((size_t)(b * Hi + yy) * Wi + xx) * (size_t)C + c];
    };
    float v00 = ld(y0c, x0c), v01 = ld(y0c, x1c);
    float v10 = ld(y1c, x0c), v11 = ld(y1c, x1c);
    float v0 = v00 + (v01 - v00) * wx;
    float v1 = v10 + (v11 - v10) * wx;
    out[i] = (_Float16)(v0 + (v1 - v0) * wy);
}

// ---------------------------------------------------------------------------
// Adaptive separable conv: out = sepconv(prev,k1v,k1h) + sepconv(next,k2v,k2h)
// Frames read f32 NCHW with edge clamp (replication pad folded into indexing).
// Kernel coefficient tensors are NHWC f16 padded to 64 channels (first 51 used).
// Output f32 NCHW.
// ---------------------------------------------------------------------------
__global__ void sepconv_kernel(const float* __restrict__ fp, const float* __restrict__ fn,
                               const _Float16* __restrict__ k1v, const _Float16* __restrict__ k1h,
                               const _Float16* __restrict__ k2v, const _Float16* __restrict__ k2h,
                               float* __restrict__ out, int Bn, int H, int W)
{
    int i = blockIdx.x * blockDim.x + threadIdx.x;
    int tot = Bn * H * W;
    if (i >= tot) return;
    const int KK = 51, PD = 25, KP = 64;
    size_t hw = (size_t)H * W;
    int b = i / (H * W);
    int r = i - b * H * W;
    int y = r / W, x = r - y * W;
    float a0 = 0.f, a1 = 0.f, a2 = 0.f;
    for (int f = 0; f < 2; ++f) {
        const float* src = (f == 0 ? fp : fn) + (size_t)b * 3 * hw;
        const _Float16* kv = (f == 0 ? k1v : k2v) + (size_t)i * KP;
        const _Float16* kh = (f == 0 ? k1h : k2h) + (size_t)i * KP;
        for (int ii = 0; ii < KK; ++ii) {
            int sy = iclampd(y + ii - PD, 0, H - 1);
            const float* row = src + (size_t)sy * W;
            float h0 = 0.f, h1 = 0.f, h2 = 0.f;
            for (int j = 0; j < KK; ++j) {
                int sx = iclampd(x + j - PD, 0, W - 1);
                float kw = (float)kh[j];
                h0 = fmaf(row[sx], kw, h0);
                h1 = fmaf(row[hw + sx], kw, h1);
                h2 = fmaf(row[2 * hw + sx], kw, h2);
            }
            float kvv = (float)kv[ii];
            a0 = fmaf(kvv, h0, a0);
            a1 = fmaf(kvv, h1, a1);
            a2 = fmaf(kvv, h2, a2);
        }
    }
    out[(size_t)b * 3 * hw + r]          = a0;
    out[(size_t)b * 3 * hw + hw + r]     = a1;
    out[(size_t)b * 3 * hw + 2 * hw + r] = a2;
}

// ---------------------------------------------------------------------------
// Host orchestration
// ---------------------------------------------------------------------------
extern "C" void kernel_launch(void* const* d_in, const int* in_sizes, int n_in,
                              void* d_out, int out_size, void* d_ws, size_t ws_size,
                              hipStream_t stream)
{
    (void)in_sizes; (void)n_in; (void)out_size; (void)ws_size;

    const int Bn = 2, H = 384, W = 384;
    const int P1 = Bn * 384 * 384;
    const int P2 = Bn * 192 * 192;
    const int P3 = Bn * 96 * 96;
    const int P4 = Bn * 48 * 48;

    struct LayerDef { int widx, cin, cout; };
    static const LayerDef L[23] = {
        { 2,   6,  32}, { 4,  32,  32},                    // e1a, e1b
        { 6,  32,  64}, { 8,  64,  64},                    // e2a, e2b
        {10,  64, 128}, {12, 128, 128},                    // e3a, e3b
        {14, 128, 256}, {16, 256, 256},                    // ma,  mb
        {18, 256, 128}, {20, 128,  64}, {22, 64, 64},      // d3, d2, d1
        {24, 64, 64}, {26, 64, 64}, {28, 64, 51},          // v1 a,b,c
        {30, 64, 64}, {32, 64, 64}, {34, 64, 51},          // h1 a,b,c
        {36, 64, 64}, {38, 64, 64}, {40, 64, 51},          // v2 a,b,c
        {42, 64, 64}, {44, 64, 64}, {46, 64, 51},          // h2 a,b,c
    };

    char* ws = (char*)d_ws;
    size_t cur = 0;
    auto alloc = [&](size_t bytes) {
        size_t o = (cur + 255) & ~(size_t)255;
        cur = o + bytes;
        return o;
    };
    auto up16 = [](int v) { return (v + 15) & ~15; };
    auto up32 = [](int v) { return (v + 31) & ~31; };

    // ---- weight arena + repack ----
    size_t woff[23];
    for (int i = 0; i < 23; ++i) {
        int cinP = up32(L[i].cin), coutP = up32(L[i].cout);
        woff[i] = alloc((size_t)9 * coutP * cinP * sizeof(_Float16));
    }
    for (int i = 0; i < 23; ++i) {
        int cinP = up32(L[i].cin), coutP = up32(L[i].cout);
        int tot = 9 * coutP * cinP;
        prep_w_kernel<<<(tot + 255) / 256, 256, 0, stream>>>(
            (const float*)d_in[L[i].widx], (_Float16*)(ws + woff[i]),
            L[i].cin, L[i].cout, cinP, coutP);
    }
    (void)up16;

    // ---- big full-res slots (reused) + small arena ----
    const size_t SLOT = (size_t)P1 * 64 * sizeof(_Float16);
    size_t G[7];
    for (int k = 0; k < 7; ++k) G[k] = alloc(SLOT);
    auto GP = [&](int k) { return (_Float16*)(ws + G[k]); };

    size_t o_p1  = alloc((size_t)P2 *  32 * 2);
    size_t o_e2t = alloc((size_t)P2 *  64 * 2);
    size_t o_e2  = alloc((size_t)P2 *  64 * 2);
    size_t o_p2  = alloc((size_t)P3 *  64 * 2);
    size_t o_e3t = alloc((size_t)P3 * 128 * 2);
    size_t o_e3  = alloc((size_t)P3 * 128 * 2);
    size_t o_p3  = alloc((size_t)P4 * 128 * 2);
    size_t o_mt  = alloc((size_t)P4 * 256 * 2);
    size_t o_m   = alloc((size_t)P4 * 256 * 2);
    size_t o_um  = alloc((size_t)P3 * 256 * 2);
    size_t o_d3  = alloc((size_t)P3 * 128 * 2);
    size_t o_ud3 = alloc((size_t)P2 * 128 * 2);
    size_t o_d2  = alloc((size_t)P2 *  64 * 2);
    auto WSH = [&](size_t o) { return (_Float16*)(ws + o); };

    auto conv = [&](const _Float16* in, int li, int Hc, int Wc, int relu,
                    const _Float16* resid, _Float16* out) {
        int cinP = up32(L[li].cin), coutP = up32(L[li].cout);
        int total = Bn * Hc * Wc;
        dim3 grid((total + 255) / 256, coutP / 32);
        conv3x3_wmma_kernel<<<grid, 256, 0, stream>>>(
            in, (const _Float16*)(ws + woff[li]),
            (const float*)d_in[L[li].widx + 1], L[li].cout,
            resid, out, total, Hc, Wc, cinP, coutP, relu);
    };
    auto pool = [&](const _Float16* in, _Float16* out, int Hi, int Wi, int C) {
        int tot = Bn * (Hi / 2) * (Wi / 2) * C;
        pool2_kernel<<<(tot + 255) / 256, 256, 0, stream>>>(in, out, Bn, Hi, Wi, C);
    };
    auto upl = [&](const _Float16* in, _Float16* out, int Hi, int Wi, int C) {
        int tot = Bn * (Hi * 2) * (Wi * 2) * C;
        up2_kernel<<<(tot + 255) / 256, 256, 0, stream>>>(in, out, Bn, Hi, Wi, C);
    };

    const float* fprev = (const float*)d_in[0];
    const float* fnext = (const float*)d_in[1];

    _Float16* x0 = GP(0);
    _Float16* t1 = GP(1);
    _Float16* e1 = GP(2);
    {
        int tot = P1 * 32;
        pack_in_kernel<<<(tot + 255) / 256, 256, 0, stream>>>(fprev, fnext, x0, H * W, Bn);
    }
    // ---- encoder ----
    conv(x0, 0, 384, 384, 1, nullptr, t1);
    conv(t1, 1, 384, 384, 1, nullptr, e1);
    pool(e1, WSH(o_p1), 384, 384, 32);
    conv(WSH(o_p1), 2, 192, 192, 1, nullptr, WSH(o_e2t));
    conv(WSH(o_e2t), 3, 192, 192, 1, nullptr, WSH(o_e2));
    pool(WSH(o_e2), WSH(o_p2), 192, 192, 64);
    conv(WSH(o_p2), 4, 96, 96, 1, nullptr, WSH(o_e3t));
    conv(WSH(o_e3t), 5, 96, 96, 1, nullptr, WSH(o_e3));
    pool(WSH(o_e3), WSH(o_p3), 96, 96, 128);
    conv(WSH(o_p3), 6, 48, 48, 1, nullptr, WSH(o_mt));
    conv(WSH(o_mt), 7, 48, 48, 1, nullptr, WSH(o_m));
    // ---- decoder with additive skips ----
    upl(WSH(o_m), WSH(o_um), 48, 48, 256);
    conv(WSH(o_um), 8, 96, 96, 1, WSH(o_e3), WSH(o_d3));
    upl(WSH(o_d3), WSH(o_ud3), 96, 96, 128);
    conv(WSH(o_ud3), 9, 192, 192, 1, WSH(o_e2), WSH(o_d2));
    _Float16* ud2 = GP(0);               // x0 dead
    upl(WSH(o_d2), ud2, 192, 192, 64);
    _Float16* f = GP(1);                 // t1 dead
    conv(ud2, 10, 384, 384, 1, nullptr, f);
    // ---- four kernel-prediction subnets ----
    _Float16* s1 = GP(2);                // e1 dead
    _Float16* s2 = GP(3);
    _Float16* kt[4] = { GP(4), GP(5), GP(6), GP(0) };   // k1v, k1h, k2v, k2h
    const int sb[4] = { 11, 14, 17, 20 };               // v1, h1, v2, h2
    for (int t = 0; t < 4; ++t) {
        conv(f, sb[t], 384, 384, 1, nullptr, s1);
        conv(s1, sb[t] + 1, 384, 384, 1, nullptr, s2);
        conv(s2, sb[t] + 2, 384, 384, 0, nullptr, kt[t]);
    }
    // ---- adaptive separable convolution ----
    sepconv_kernel<<<(P1 + 255) / 256, 256, 0, stream>>>(
        fprev, fnext, kt[0], kt[1], kt[2], kt[3], (float*)d_out, Bn, H, W);
}
